// ManifoldCtrlInvLoss_29257317220643
// MI455X (gfx1250) — matrix operand order repaired and checked by reading, not compiled, not run on hardware
//
#include <hip/hip_runtime.h>
#include <math.h>

// ---------------- problem constants (from reference) ----------------
#define DIMS        128
#define KNN         16
#define EPSF        1e-8f
#define IDXMASK     0x1FFFu          // 13 bits: n = 8192
#define KEYMASK     0xFFFFE000u

// ---------------- knn kernel geometry ----------------
#define CTILE        16              // columns per WMMA tile
#define ROWS_PER_WAVE 32             // two stacked 16x16 C tiles
#define WAVES_PER_BLOCK 8
#define ROWS_PER_BLOCK (ROWS_PER_WAVE * WAVES_PER_BLOCK)  // 256
#define C_SPLIT      8               // column-dimension split for parallelism
#define KSLABS       (DIMS / 32)     // 4 K-slabs of 32 for bf16 WMMA

typedef __attribute__((ext_vector_type(8)))  float   v8f;
typedef __attribute__((ext_vector_type(4)))  __bf16  v4bf;
typedef __attribute__((ext_vector_type(8)))  __bf16  v8bf;
typedef __attribute__((ext_vector_type(16))) __bf16  v16bf;

// -------------------------------------------------------------------
// Kernel 0: fp32 -> bf16 conversion of U_real (ranking operand only)
// -------------------------------------------------------------------
__global__ void cvt_kernel(const float* __restrict__ X, __bf16* __restrict__ Xh,
                           int total4) {
    int i = blockIdx.x * blockDim.x + threadIdx.x;
    if (i >= total4) return;
    float4 v = ((const float4*)X)[i];
    v4bf o;
    o[0] = (__bf16)v.x; o[1] = (__bf16)v.y; o[2] = (__bf16)v.z; o[3] = (__bf16)v.w;
    *(v4bf*)(Xh + (size_t)i * 4) = o;
}

// -------------------------------------------------------------------
// Kernel 1: row squared norms (exact fp32).  One wave per row.
// -------------------------------------------------------------------
__global__ void sq_kernel(const float* __restrict__ X, float* __restrict__ sq, int n) {
    int wave = threadIdx.x >> 5;
    int lane = threadIdx.x & 31;
    int row  = blockIdx.x * 8 + wave;
    if (row >= n) return;
    const float4* xr = (const float4*)(X + (size_t)row * DIMS);
    float4 v = xr[lane];                       // 32 lanes x 4 floats = 128
    float s = v.x * v.x + v.y * v.y + v.z * v.z + v.w * v.w;
    #pragma unroll
    for (int m = 16; m >= 1; m >>= 1) s += __shfl_xor(s, m, 32);
    if (lane == 0) sq[row] = s;
}

// -------------------------------------------------------------------
// Kernel 2: fused Gram (WMMA f32_16x16x32_bf16) + per-row top-16.
// grid = (n / ROWS_PER_BLOCK, C_SPLIT), block = 256 threads (8 waves).
// Each wave: 32 rows x (n/C_SPLIT) columns; A preloaded in registers.
// Top-16 kept as packed u32 keys: (d2 bits & KEYMASK) | column-index.
// -------------------------------------------------------------------
__global__ void __launch_bounds__(256, 1)
knn_wmma_kernel(const __bf16* __restrict__ Xh,
                const float*  __restrict__ sqn,
                unsigned int* __restrict__ pkeys,
                int n) {
    __shared__ __bf16 ldsBh[CTILE * DIMS];                            // 4 KB
    __shared__ float  ldsC[WAVES_PER_BLOCK * ROWS_PER_WAVE * CTILE];  // 16 KB

    const int t      = threadIdx.x;
    const int wave   = t >> 5;
    const int lane   = t & 31;
    const int lane15 = lane & 15;
    const int hi     = lane >> 4;                // 0 or 1

    const int rowBase   = blockIdx.x * ROWS_PER_BLOCK + wave * ROWS_PER_WAVE;
    const int seg       = blockIdx.y;
    const int segCols   = n / C_SPLIT;           // 1024
    const int segBase   = seg * segCols;
    const int tilesSeg  = segCols / CTILE;       // 64

    // ---- preload A fragments for this wave's 32 rows (64 VGPRs) ----
    // 16-bit A 16x32 layout: lane<16 -> K 0..7 (v0-3), 16..23 (v4-7);
    //                        lane>=16 -> K 8..15, 24..31.
    v16bf a0s[KSLABS], a1s[KSLABS];
    {
        const __bf16* arow0 = Xh + (size_t)(rowBase + lane15) * DIMS;
        const __bf16* arow1 = arow0 + (size_t)16 * DIMS;
        #pragma unroll
        for (int s = 0; s < KSLABS; ++s) {
            const int o0 = s * 32 + 8 * hi;      // K-slab base + lane-half
            v8bf lo0 = *(const v8bf*)(arow0 + o0);
            v8bf hi0 = *(const v8bf*)(arow0 + o0 + 16);
            a0s[s] = __builtin_shufflevector(lo0, hi0, 0,1,2,3,4,5,6,7,
                                                       8,9,10,11,12,13,14,15);
            v8bf lo1 = *(const v8bf*)(arow1 + o0);
            v8bf hi1 = *(const v8bf*)(arow1 + o0 + 16);
            a1s[s] = __builtin_shufflevector(lo1, hi1, 0,1,2,3,4,5,6,7,
                                                       8,9,10,11,12,13,14,15);
        }
    }

    // This lane owns row (rowBase + lane) in the scan/top-k phase.
    const int   selfIdx = rowBase + lane;
    const float sqRow   = sqn[selfIdx];

    float* myC = ldsC + wave * (ROWS_PER_WAVE * CTILE);

    // register-resident sorted top-16 of packed keys (ascending)
    unsigned int kd[KNN];
    #pragma unroll
    for (int s = 0; s < KNN; ++s) kd[s] = 0xFFFFFFFFu;

    for (int ct = 0; ct < tilesSeg; ++ct) {
        const int colBase = segBase + ct * CTILE;

        __syncthreads();  // protect ldsBh from previous iteration's readers
        {   // cooperative load of B tile: 16 contiguous bf16 rows -> 4 KB
            const uint4* bsrc = (const uint4*)(Xh + (size_t)colBase * DIMS);
            uint4* bdst = (uint4*)ldsBh;
            bdst[t] = bsrc[t];                   // 256 threads x 16 B = 4 KB
            if (ct + 1 < tilesSeg) {             // prefetch next B tile (near caches)
                const char* nb = (const char*)(Xh + (size_t)(colBase + CTILE) * DIMS);
                __builtin_prefetch(nb + t * 16, 0, 3);
            }
        }
        __syncthreads();

        // ---- GEMM: two 16x16 C tiles, K = 128 as 4 slabs of 32 ----
        // B 32x16 layout: lane<16 -> K 0..7, 16..23; lane>=16 -> K 8..15, 24..31.
        v8f c0 = {};
        v8f c1 = {};
        const __bf16* bcol = ldsBh + lane15 * DIMS;
        #pragma unroll
        for (int s = 0; s < KSLABS; ++s) {
            const __bf16* bp = bcol + s * 32 + 8 * hi;
            v8bf blo = *(const v8bf*)(bp);
            v8bf bhi = *(const v8bf*)(bp + 16);
            v16bf b = __builtin_shufflevector(blo, bhi, 0,1,2,3,4,5,6,7,
                                                        8,9,10,11,12,13,14,15);
            c0 = __builtin_amdgcn_wmma_f32_16x16x32_bf16(false, a0s[s], false, b,
                                                         (short)0, c0, false, false);
            c1 = __builtin_amdgcn_wmma_f32_16x16x32_bf16(false, a1s[s], false, b,
                                                         (short)0, c1, false, false);
        }

        // ---- transpose via per-wave LDS scratch: store sq_col - 2*dot ----
        const float sqc = sqn[colBase + lane15];
        #pragma unroll
        for (int r = 0; r < 8; ++r) {
            myC[(r + 8 * hi) * CTILE + lane15]      = sqc - 2.0f * c0[r];
            myC[(16 + r + 8 * hi) * CTILE + lane15] = sqc - 2.0f * c1[r];
        }
        asm volatile("s_wait_dscnt 0" ::: "memory");  // wave-internal LDS RAW

        // ---- per-row top-16 update: lane owns row `lane` of the 32 ----
        const float4* rowC = (const float4*)(myC + lane * CTILE);
        #pragma unroll
        for (int j4 = 0; j4 < 4; ++j4) {
            float4 v = rowC[j4];
            float cv4[4] = {v.x, v.y, v.z, v.w};
            #pragma unroll
            for (int e = 0; e < 4; ++e) {
                float d2 = fmaxf(sqRow + cv4[e], 0.0f);   // reference clamps too
                int   ci = colBase + j4 * 4 + e;
                unsigned int key = (__float_as_uint(d2) & KEYMASK) | (unsigned int)ci;
                if (key < kd[KNN - 1] && ci != selfIdx) {
                    unsigned int cv = key;
                    #pragma unroll
                    for (int s = 0; s < KNN; ++s) {
                        unsigned int lo = (cv < kd[s]) ? cv : kd[s];
                        cv    = (cv < kd[s]) ? kd[s] : cv;
                        kd[s] = lo;
                    }
                }
            }
        }
    }

    // write partial key list for this (row, segment)
    const size_t base = (size_t)selfIdx * (C_SPLIT * KNN) + (size_t)seg * KNN;
    #pragma unroll
    for (int s = 0; s < KNN; ++s) pkeys[base + s] = kd[s];
}

// -------------------------------------------------------------------
// Kernel 3: merge C_SPLIT partial top-16 key lists -> final indices/row
// -------------------------------------------------------------------
__global__ void merge_kernel(const unsigned int* __restrict__ pkeys,
                             int* __restrict__ knn, int n) {
    int row = blockIdx.x * blockDim.x + threadIdx.x;
    if (row >= n) return;
    unsigned int kd[KNN];
    #pragma unroll
    for (int s = 0; s < KNN; ++s) kd[s] = 0xFFFFFFFFu;
    const unsigned int* pk = pkeys + (size_t)row * (C_SPLIT * KNN);
    for (int c = 0; c < C_SPLIT * KNN; ++c) {
        unsigned int cv = pk[c];
        if (cv < kd[KNN - 1]) {
            #pragma unroll
            for (int s = 0; s < KNN; ++s) {
                unsigned int lo = (cv < kd[s]) ? cv : kd[s];
                cv    = (cv < kd[s]) ? kd[s] : cv;
                kd[s] = lo;
            }
        }
    }
    #pragma unroll
    for (int s = 0; s < KNN; ++s)
        knn[(size_t)row * KNN + s] = (int)(kd[s] & IDXMASK);
}

// -------------------------------------------------------------------
// Kernel 4: zero the output scalar
// -------------------------------------------------------------------
__global__ void init_out_kernel(float* out) { out[0] = 0.0f; }

// -------------------------------------------------------------------
// Kernel 5: loss (exact fp32).  One block (128 threads = 4 waves) per row.
// -------------------------------------------------------------------
__global__ void loss_kernel(const float* __restrict__ rec,
                            const float* __restrict__ real,
                            const int*   __restrict__ knn,
                            float* __restrict__ out, int n) {
    const int i    = blockIdx.x;
    const int t    = threadIdx.x;
    const int wave = t >> 5;
    const int lane = t & 31;

    __shared__ float s_dr[KNN];
    __shared__ float s_dc[KNN];
    __shared__ float s_l1[4];

    // loss1 partial: (rec - real)^2 for element (i, t)
    {
        float dif = rec[(size_t)i * DIMS + t] - real[(size_t)i * DIMS + t];
        float l1 = dif * dif;
        #pragma unroll
        for (int m = 16; m >= 1; m >>= 1) l1 += __shfl_xor(l1, m, 32);
        if (lane == 0) s_l1[wave] = l1;
    }

    // neighbor distances: wave w handles k = 4w..4w+3
    #pragma unroll
    for (int kk = 0; kk < 4; ++kk) {
        int k = wave * 4 + kk;
        int j = knn[(size_t)i * KNN + k];
        float ar = 0.f, ac = 0.f;
        #pragma unroll
        for (int c = 0; c < 4; ++c) {
            int d = lane + c * 32;
            float dr = real[(size_t)i * DIMS + d] - real[(size_t)j * DIMS + d];
            float dc = rec [(size_t)i * DIMS + d] - rec [(size_t)j * DIMS + d];
            ar += dr * dr;
            ac += dc * dc;
        }
        #pragma unroll
        for (int m = 16; m >= 1; m >>= 1) {
            ar += __shfl_xor(ar, m, 32);
            ac += __shfl_xor(ac, m, 32);
        }
        if (lane == 0) { s_dr[k] = sqrtf(ar); s_dc[k] = sqrtf(ac); }
    }
    __syncthreads();

    if (t < 32) {
        float dr = (lane < KNN) ? s_dr[lane] : 0.f;
        float dc = (lane < KNN) ? s_dc[lane] : 0.f;
        float mr = dr, mc = dc;
        #pragma unroll
        for (int m = 16; m >= 1; m >>= 1) {
            mr = fmaxf(mr, __shfl_xor(mr, m, 32));
            mc = fmaxf(mc, __shfl_xor(mc, m, 32));
        }
        float term = (lane < KNN) ? fabsf(dr / (mr + EPSF) - dc / (mc + EPSF)) : 0.f;
        #pragma unroll
        for (int m = 16; m >= 1; m >>= 1) term += __shfl_xor(term, m, 32);
        if (lane == 0) {
            float l1 = s_l1[0] + s_l1[1] + s_l1[2] + s_l1[3];
            float contrib = l1 / ((float)n * (float)DIMS)
                          + term / ((float)n * (float)KNN);
            atomicAdd(out, contrib);
        }
    }
}

// -------------------------------------------------------------------
extern "C" void kernel_launch(void* const* d_in, const int* in_sizes, int n_in,
                              void* d_out, int out_size, void* d_ws, size_t ws_size,
                              hipStream_t stream) {
    const float* U_recover = (const float*)d_in[0];
    const float* U_real    = (const float*)d_in[1];
    float* out = (float*)d_out;
    const int n = in_sizes[0] / DIMS;   // 8192

    // workspace layout
    char* ws = (char*)d_ws;
    __bf16* Xh  = (__bf16*)ws;                   ws += (size_t)n * DIMS * sizeof(__bf16);
    float* sq    = (float*)ws;                   ws += (size_t)n * sizeof(float);
    unsigned int* pkeys = (unsigned int*)ws;     ws += (size_t)n * C_SPLIT * KNN * sizeof(unsigned int);
    int*   knn   = (int*)ws;

    // 0) bf16 copy of U_real for the ranking GEMM
    const int total4 = n * DIMS / 4;
    cvt_kernel<<<dim3((total4 + 255) / 256), dim3(256), 0, stream>>>(U_real, Xh, total4);

    // 1) exact fp32 row norms of U_real
    sq_kernel<<<dim3((n + 7) / 8), dim3(256), 0, stream>>>(U_real, sq, n);

    // 2) fused bf16 WMMA Gram + packed top-16 per (row, column-segment)
    dim3 g2(n / ROWS_PER_BLOCK, C_SPLIT);
    knn_wmma_kernel<<<g2, dim3(256), 0, stream>>>(Xh, sq, pkeys, n);

    // 3) merge partials -> final neighbor indices
    merge_kernel<<<dim3((n + 255) / 256), dim3(256), 0, stream>>>(pkeys, knn, n);

    // 4) zero output, 5) exact fp32 loss
    init_out_kernel<<<dim3(1), dim3(1), 0, stream>>>(out);
    loss_kernel<<<dim3(n), dim3(128), 0, stream>>>(U_recover, U_real, knn, out, n);
}
